// CrossMLA_89000312308236
// MI455X (gfx1250) — matrix operand order repaired and checked by reading, not compiled, not run on hardware
//
#include <hip/hip_runtime.h>

typedef __bf16 bf16_t;
typedef __bf16 v16bf __attribute__((ext_vector_type(16)));
typedef __bf16 v8bf  __attribute__((ext_vector_type(8)));
typedef float  v8f   __attribute__((ext_vector_type(8)));
typedef unsigned int       u32;
typedef unsigned long long u64;
typedef u32 v4u __attribute__((ext_vector_type(4)));
typedef int  v4i __attribute__((ext_vector_type(4)));
typedef int  v8i __attribute__((ext_vector_type(8)));

#define DEV static __device__ __forceinline__

// ---------------------------------------------------------------------------
// WMMA helpers (CDNA5 V_WMMA_F32_16X16X32_BF16, wave32)
// ---------------------------------------------------------------------------
DEV v8f wmma_bf16(v16bf a, v16bf b, v8f c) {
  return __builtin_amdgcn_wmma_f32_16x16x32_bf16(false, a, false, b, (short)0, c,
                                                 false, false);
}

// A-matrix 16x32 bf16 fragment (ISA 7.12.2):
//   lanes 0-15 : row M=lane,    elements {K0..K7,  K16..K23}
//   lanes 16-31: row M=lane-16, elements {K8..K15, K24..K31}
DEV v16bf a_frag_from(const bf16_t* p /* row base + kb */) {
  v8bf lo = *(const v8bf*)p;
  v8bf hi = *(const v8bf*)(p + 16);
  v16bf r;
#pragma unroll
  for (int i = 0; i < 8; ++i) { r[i] = lo[i]; r[i + 8] = hi[i]; }
  return r;
}
// B-matrix 32x16 fragment from BT[N][K] row-major:
//   lanes 0-15 : col N=lane,    elements {K0..K7,  K16..K23}
//   lanes 16-31: col N=lane-16, elements {K8..K15, K24..K31}
DEV v16bf b_frag_from(const bf16_t* p /* col base + kk */) {
  v8bf lo = *(const v8bf*)p;
  v8bf hi = *(const v8bf*)(p + 16);
  v16bf r;
#pragma unroll
  for (int i = 0; i < 8; ++i) { r[i] = lo[i]; r[i + 8] = hi[i]; }
  return r;
}

DEV v16bf load_a_frag(const bf16_t* A, int lda, int m0, int k0) {
  int lane = threadIdx.x & 31;
  return a_frag_from(A + (size_t)(m0 + (lane & 15)) * lda + k0 +
                     ((lane & 16) ? 8 : 0));
}

DEV float half_max(float v) {
  v = fmaxf(v, __shfl_xor(v, 1, 32));
  v = fmaxf(v, __shfl_xor(v, 2, 32));
  v = fmaxf(v, __shfl_xor(v, 4, 32));
  v = fmaxf(v, __shfl_xor(v, 8, 32));
  return v;
}
DEV float half_sum(float v) {
  v += __shfl_xor(v, 1, 32);
  v += __shfl_xor(v, 2, 32);
  v += __shfl_xor(v, 4, 32);
  v += __shfl_xor(v, 8, 32);
  return v;
}

// ---------------------------------------------------------------------------
// Tensor Data Mover: 2D tile DMA global -> LDS (D# per ISA 8.3/8.4)
// 6-arg builtin form: (g0:u32x4, g1:i32x8, g2:i32x4, g3:i32x4, g1ext:i32x8, cpol)
// ---------------------------------------------------------------------------
DEV void tdm_load_2d(u32 lds_off, const bf16_t* gptr,
                     u32 tile_k /* elems, dim0 */, u32 tile_rows /* dim1 */,
                     u32 tensor_d0 /* elems */, u64 stride_elems,
                     u32 tensor_d1) {
  u64 ga = (u64)(uintptr_t)gptr;
  v4u g0;
  g0[0] = 1u;                                               // count=1, user
  g0[1] = lds_off;                                          // lds_addr (bytes)
  g0[2] = (u32)ga;                                          // global_addr lo
  g0[3] = (u32)((ga >> 32) & 0x1FFFFFFull) | (2u << 30);    // addr hi | type=2
  // group1 (256 bits) assembled as four u64 words
  u64 q0 = (1ull << 16)                                     // data_size = 2B
         | (((u64)(tensor_d0 & 0xFFFFu)) << 48);            // tensor_dim0[15:0]
  u64 q1 = (u64)(tensor_d0 >> 16)                           // tensor_dim0[31:16]
         | ((u64)tensor_d1 << 16)                           // tensor_dim1
         | ((u64)tile_k << 48);                             // tile_dim0
  u64 q2 = (u64)tile_rows                                   // tile_dim1
         | ((stride_elems & 0xFFFFFFFFull) << 32);          // d0_stride[31:0]
  u64 q3 = (stride_elems >> 32);                            // d0_stride[47:32]
  v8i g1;
  g1[0] = (int)(u32)q0; g1[1] = (int)(u32)(q0 >> 32);
  g1[2] = (int)(u32)q1; g1[3] = (int)(u32)(q1 >> 32);
  g1[4] = (int)(u32)q2; g1[5] = (int)(u32)(q2 >> 32);
  g1[6] = (int)(u32)q3; g1[7] = (int)(u32)(q3 >> 32);
  v4i z4 = {0, 0, 0, 0};
  v8i z8 = {0, 0, 0, 0, 0, 0, 0, 0};
  __builtin_amdgcn_tensor_load_to_lds(g0, g1, z4, z4, z8, 0);
}

// ---------------------------------------------------------------------------
// Elementwise / prep kernels
// ---------------------------------------------------------------------------
__global__ void k_f32_to_bf16(const float* __restrict__ src,
                              bf16_t* __restrict__ dst, size_t n) {
  size_t i = blockIdx.x * (size_t)blockDim.x + threadIdx.x;
  if (i < n) dst[i] = (bf16_t)src[i];
}

// src[K][N] f32 -> dst[Np][Kp] bf16, zero padded (pre-transposed B for WMMA)
__global__ void k_transpose_bf16(const float* __restrict__ src,
                                 bf16_t* __restrict__ dst,
                                 int K, int N, int Kp, int Np) {
  int n = blockIdx.x * blockDim.x + threadIdx.x;
  int k = blockIdx.y * blockDim.y + threadIdx.y;
  if (n < Np && k < Kp) {
    float v = (n < N && k < K) ? src[(size_t)k * N + n] : 0.0f;
    dst[(size_t)n * Kp + k] = (bf16_t)v;
  }
}

__global__ void k_layernorm_bf16(const float* __restrict__ src, int srcld,
                                 bf16_t* __restrict__ dst, int dstld,
                                 int W, int Wp) {
  __shared__ float red[256];
  int row = blockIdx.x;
  const float* x = src + (size_t)row * srcld;
  float s = 0.f, s2 = 0.f;
  for (int i = threadIdx.x; i < W; i += blockDim.x) {
    float v = x[i];
    s += v; s2 += v * v;
  }
  red[threadIdx.x] = s;  __syncthreads();
  for (int o = blockDim.x >> 1; o > 0; o >>= 1) {
    if (threadIdx.x < (unsigned)o) red[threadIdx.x] += red[threadIdx.x + o];
    __syncthreads();
  }
  float mean = red[0] / W;
  __syncthreads();
  red[threadIdx.x] = s2; __syncthreads();
  for (int o = blockDim.x >> 1; o > 0; o >>= 1) {
    if (threadIdx.x < (unsigned)o) red[threadIdx.x] += red[threadIdx.x + o];
    __syncthreads();
  }
  float var = red[0] / W - mean * mean;
  float inv = rsqrtf(var + 1e-5f);
  for (int i = threadIdx.x; i < Wp; i += blockDim.x) {
    float v = (i < W) ? (x[i] - mean) * inv : 0.0f;
    dst[(size_t)row * dstld + i] = (bf16_t)v;
  }
}

__global__ void k_assemble_q(const float* __restrict__ Qf,
                             const float* __restrict__ cosc,
                             const float* __restrict__ sinc,
                             bf16_t* __restrict__ qh) {
  size_t idx = blockIdx.x * (size_t)blockDim.x + threadIdx.x;
  if (idx >= 4096ull * 2048ull) return;
  int col = (int)(idx & 2047);
  int row = (int)(idx >> 11);
  int h = col >> 7, d = col & 127;
  int b = row >> 11, s = row & 2047;
  const float* q = Qf + (size_t)row * 2048 + h * 128;
  float v;
  if (d < 64) {
    v = q[d];
  } else {
    int j = d - 64;
    float c  = cosc[(size_t)s * 64 + (j >> 1)];
    float sn = sinc[(size_t)s * 64 + (j >> 1)];
    float x  = q[64 + j];
    float xr = (j < 32) ? -q[64 + j + 32] : q[64 + j - 32];
    v = x * c + xr * sn;
  }
  qh[((size_t)(b * 16 + h) * 2048 + s) * 128 + d] =
      (bf16_t)(v * 0.08838834764831845f);
}

__global__ void k_assemble_kv(const float* __restrict__ KVf,
                              const float* __restrict__ Ckv,
                              const float* __restrict__ cosc,
                              const float* __restrict__ sinc,
                              bf16_t* __restrict__ kh,
                              bf16_t* __restrict__ vt) {
  size_t idx = blockIdx.x * (size_t)blockDim.x + threadIdx.x;
  const size_t total = 4096ull * 16ull * 192ull;
  if (idx >= total) return;
  int e = (int)(idx % 192);
  size_t t1 = idx / 192;
  int h = (int)(t1 % 16);
  int row = (int)(t1 / 16);
  int b = row >> 11, s = row & 2047;
  if (e < 64) {
    float v = KVf[(size_t)row * 3072 + h * 192 + e];
    kh[((size_t)(b * 16 + h) * 2048 + s) * 128 + e] = (bf16_t)v;
  } else if (e < 128) {
    int j = e - 64;
    const float* kr = Ckv + (size_t)row * 1536 + 1365;
    float c  = cosc[(size_t)s * 64 + (j >> 1)];
    float sn = sinc[(size_t)s * 64 + (j >> 1)];
    float x  = kr[j];
    float xr = (j < 32) ? -kr[j + 32] : kr[j - 32];
    kh[((size_t)(b * 16 + h) * 2048 + s) * 128 + e] = (bf16_t)(x * c + xr * sn);
  } else {
    int dv = e - 128;
    float v = KVf[(size_t)row * 3072 + h * 192 + 64 + dv];
    vt[((size_t)(b * 16 + h) * 128 + dv) * 2048 + s] = (bf16_t)v;
  }
}

__global__ void k_copy_ckv(const float* __restrict__ Ckv,
                           float* __restrict__ out) {
  size_t idx = blockIdx.x * (size_t)blockDim.x + threadIdx.x;
  const size_t total = 4096ull * 1429ull;
  if (idx >= total) return;
  size_t r = idx / 1429;
  int c = (int)(idx % 1429);
  out[idx] = Ckv[r * 1536 + c];
}

// ---------------------------------------------------------------------------
// TDM-fed, LDS double-buffered bf16 WMMA GEMM.
// C[M,N] = A[M,K] * BT[N,K]^T, f32 accumulate.
// Block: 256 threads = 8 waves (2 x 4). Block tile 128x128, K-step 64.
// Wave 0 drives the Tensor Data Mover; TENSORcnt + barrier synchronize.
// ---------------------------------------------------------------------------
__global__ void __launch_bounds__(256)
k_gemm_bf16(const bf16_t* __restrict__ A, const bf16_t* __restrict__ BT,
            float* __restrict__ C, int M, int N, int K,
            int lda, int ldb, int ldc) {
  __shared__ bf16_t As[2][128][64];
  __shared__ bf16_t Bs[2][128][64];
  int wave = threadIdx.x >> 5;
  int lane = threadIdx.x & 31;
  int wm = wave >> 2;   // 0..1 -> 64-row strip
  int wn = wave & 3;    // 0..3 -> 32-col strip
  int m0 = blockIdx.y * 128;
  int n0 = blockIdx.x * 128;
  bool issuer = (wave == 0);

  int nk = K >> 6;
  if (issuer) {
    tdm_load_2d((u32)(uintptr_t)&As[0][0][0], A + (size_t)m0 * lda,
                64, 128, (u32)K, (u64)lda, (u32)M);
    tdm_load_2d((u32)(uintptr_t)&Bs[0][0][0], BT + (size_t)n0 * ldb,
                64, 128, (u32)K, (u64)ldb, (u32)N);
  }

  v8f acc[4][2];
#pragma unroll
  for (int mt = 0; mt < 4; ++mt)
#pragma unroll
    for (int nt = 0; nt < 2; ++nt)
#pragma unroll
      for (int r = 0; r < 8; ++r) acc[mt][nt][r] = 0.0f;

  int la = lane & 15;
  int kbA = (lane & 16) ? 8 : 0;
  int kkB = (lane & 16) ? 8 : 0;

  for (int ks = 0; ks < nk; ++ks) {
    int cur = ks & 1;
    if (issuer) __builtin_amdgcn_s_wait_tensorcnt((short)0);
    __syncthreads();                       // buffers[cur] ready for all waves
    if (issuer && (ks + 1 < nk)) {         // prefetch next slab via TDM
      int k0 = (ks + 1) << 6;
      tdm_load_2d((u32)(uintptr_t)&As[cur ^ 1][0][0],
                  A + (size_t)m0 * lda + k0, 64, 128, (u32)(K - k0),
                  (u64)lda, (u32)M);
      tdm_load_2d((u32)(uintptr_t)&Bs[cur ^ 1][0][0],
                  BT + (size_t)n0 * ldb + k0, 64, 128, (u32)(K - k0),
                  (u64)ldb, (u32)N);
    }
#pragma unroll
    for (int ksub = 0; ksub < 64; ksub += 32) {
      v16bf bfrag[2];
#pragma unroll
      for (int nt = 0; nt < 2; ++nt)
        bfrag[nt] = b_frag_from(&Bs[cur][wn * 32 + nt * 16 + la][ksub + kkB]);
#pragma unroll
      for (int mt = 0; mt < 4; ++mt) {
        v16bf afrag = a_frag_from(&As[cur][wm * 64 + mt * 16 + la][ksub + kbA]);
        acc[mt][0] = wmma_bf16(afrag, bfrag[0], acc[mt][0]);
        acc[mt][1] = wmma_bf16(afrag, bfrag[1], acc[mt][1]);
      }
    }
    __syncthreads();                       // all reads done before overwrite
  }

  int mh = (lane & 16) ? 8 : 0;
#pragma unroll
  for (int mt = 0; mt < 4; ++mt)
#pragma unroll
    for (int nt = 0; nt < 2; ++nt)
#pragma unroll
      for (int r = 0; r < 8; ++r) {
        int m = m0 + wm * 64 + mt * 16 + mh + r;
        int nn = n0 + wn * 32 + nt * 16 + la;
        if (m < M && nn < N) C[(size_t)m * ldc + nn] = acc[mt][nt][r];
      }
}

// ---------------------------------------------------------------------------
// TDM-fed flash attention.  Block = 8 waves sharing one (b,h); each wave owns
// 16 query rows.  K/V tiles (kv-width 64) are DMA'd into double-buffered LDS
// by wave 0 (all 8 waves reuse them); online softmax; P staged via LDS to
// convert C-layout -> A-layout; both Q·K^T and P·V on WMMA.
// qh/kh: [B,H,S,128] bf16.  vt: [B,H,128,S] bf16.
// ---------------------------------------------------------------------------
__global__ void __launch_bounds__(256)
k_flash_attn(const bf16_t* __restrict__ qh,
             const bf16_t* __restrict__ kh,
             const bf16_t* __restrict__ vt,
             bf16_t* __restrict__ attn) {
  __shared__ bf16_t Ks[2][64][128];   // kv rows, d contiguous
  __shared__ bf16_t Vs[2][128][64];   // d rows, kv contiguous
  __shared__ bf16_t Plds[8][16][72];  // per-wave 16x64 P tile (padded stride)
  int wave = threadIdx.x >> 5;
  int lane = threadIdx.x & 31;
  int b = blockIdx.z, h = blockIdx.y;
  int q0 = blockIdx.x * 128 + wave * 16;
  const bf16_t* Q = qh + (size_t)(b * 16 + h) * 2048 * 128;
  const bf16_t* K = kh + (size_t)(b * 16 + h) * 2048 * 128;
  const bf16_t* V = vt + (size_t)(b * 16 + h) * 128 * 2048;
  bool issuer = (wave == 0);

  v16bf qa[4];
#pragma unroll
  for (int t = 0; t < 4; ++t) qa[t] = load_a_frag(Q, 128, q0, t * 32);

  if (issuer) {
    tdm_load_2d((u32)(uintptr_t)&Ks[0][0][0], K, 128, 64, 128, 128, 2048);
    tdm_load_2d((u32)(uintptr_t)&Vs[0][0][0], V, 64, 128, 2048, 2048, 128);
  }

  float mrow[8], lrow[8];
  v8f o[8];
#pragma unroll
  for (int r = 0; r < 8; ++r) { mrow[r] = -1e30f; lrow[r] = 0.0f; }
#pragma unroll
  for (int t = 0; t < 8; ++t)
#pragma unroll
    for (int r = 0; r < 8; ++r) o[t][r] = 0.0f;

  int mh  = (lane & 16) ? 8 : 0;
  int la  = lane & 15;
  int kbA = (lane & 16) ? 8 : 0;
  int kkB = (lane & 16) ? 8 : 0;

  for (int ks = 0; ks < 32; ++ks) {   // 2048 kv in steps of 64
    int cur = ks & 1;
    if (issuer) __builtin_amdgcn_s_wait_tensorcnt((short)0);
    __syncthreads();                  // K/V tile[cur] ready for all waves
    if (issuer && (ks + 1 < 32)) {    // TDM prefetch next kv tile
      int kv0 = (ks + 1) * 64;
      tdm_load_2d((u32)(uintptr_t)&Ks[cur ^ 1][0][0], K + (size_t)kv0 * 128,
                  128, 64, 128, 128, (u32)(2048 - kv0));
      tdm_load_2d((u32)(uintptr_t)&Vs[cur ^ 1][0][0], V + kv0,
                  64, 128, (u32)(2048 - kv0), 2048, 128);
    }
    // ---- S = Q * K^T : 16 x 64 tile = 4 C fragments ----
    v8f s[4];
#pragma unroll
    for (int nt = 0; nt < 4; ++nt)
#pragma unroll
      for (int r = 0; r < 8; ++r) s[nt][r] = 0.0f;
#pragma unroll
    for (int t = 0; t < 4; ++t)
#pragma unroll
      for (int nt = 0; nt < 4; ++nt) {
        v16bf bf = b_frag_from(&Ks[cur][nt * 16 + la][t * 32 + kkB]);
        s[nt] = wmma_bf16(qa[t], bf, s[nt]);
      }
    // ---- online softmax stats (rows live in 16-lane halves) ----
    float alpha[8];
#pragma unroll
    for (int r = 0; r < 8; ++r) {
      float nm = fmaxf(fmaxf(s[0][r], s[1][r]), fmaxf(s[2][r], s[3][r]));
      nm = half_max(nm);
      float m2 = fmaxf(mrow[r], nm);
      alpha[r] = __expf(mrow[r] - m2);
      mrow[r] = m2;
    }
#pragma unroll
    for (int r = 0; r < 8; ++r) {
      float sum = 0.0f;
#pragma unroll
      for (int nt = 0; nt < 4; ++nt) {
        float p = __expf(s[nt][r] - mrow[r]);
        Plds[wave][mh + r][nt * 16 + la] = (bf16_t)p;
        sum += p;
      }
      lrow[r] = lrow[r] * alpha[r] + half_sum(sum);
    }
    __syncthreads();                  // P tile visible for A-layout reload
#pragma unroll
    for (int t = 0; t < 8; ++t)
#pragma unroll
      for (int r = 0; r < 8; ++r) o[t][r] *= alpha[r];
    // ---- O += P * V over two 32-wide kv substeps ----
#pragma unroll
    for (int ksub = 0; ksub < 64; ksub += 32) {
      v16bf pa;
#pragma unroll
      for (int i = 0; i < 8; ++i) {
        pa[i]     = Plds[wave][la][ksub + kbA + i];
        pa[i + 8] = Plds[wave][la][ksub + kbA + 16 + i];
      }
#pragma unroll
      for (int t = 0; t < 8; ++t) {
        v16bf vb = b_frag_from(&Vs[cur][t * 16 + la][ksub + kkB]);
        o[t] = wmma_bf16(pa, vb, o[t]);
      }
    }
    __syncthreads();                  // reads of tile[cur]/P done before reuse
  }
  // ---- epilogue: normalize, store to [b, s, h*128+d] bf16 ----
#pragma unroll
  for (int t = 0; t < 8; ++t)
#pragma unroll
    for (int r = 0; r < 8; ++r) {
      int s = q0 + mh + r;
      int d = t * 16 + la;
      float val = o[t][r] / lrow[r];
      attn[((size_t)(b * 2048 + s)) * 2048 + h * 128 + d] = (bf16_t)val;
    }
}

// ---------------------------------------------------------------------------
// Host launcher
// ---------------------------------------------------------------------------
extern "C" void kernel_launch(void* const* d_in, const int* in_sizes, int n_in,
                              void* d_out, int out_size, void* d_ws,
                              size_t ws_size, hipStream_t stream) {
  (void)in_sizes; (void)n_in; (void)out_size; (void)ws_size;

  const float* q_input  = (const float*)d_in[0];
  const float* kv_input = (const float*)d_in[1];
  const float* W_dq     = (const float*)d_in[2];
  const float* W_uq     = (const float*)d_in[3];
  const float* W_dkv    = (const float*)d_in[4];
  const float* W_ukv    = (const float*)d_in[5];
  const float* W_o      = (const float*)d_in[6];
  const float* cosc     = (const float*)d_in[7];
  const float* sinc     = (const float*)d_in[8];

  float* out     = (float*)d_out;   // [2,2048,2048]
  float* ckv_out = out + 8388608;   // [2,2048,1429]

  char* ws = (char*)d_ws;
  size_t off = 0;
  auto carve = [&](size_t bytes) -> char* {
    char* p = ws + off;
    off = (off + bytes + 255) & ~(size_t)255;
    return p;
  };
  const size_t R = 4096;  // B * S
  bf16_t* q_in   = (bf16_t*)carve(R * 2048 * 2);
  bf16_t* kv_in  = (bf16_t*)carve(R * 2048 * 2);
  bf16_t* Wdq_t  = (bf16_t*)carve((size_t)1024 * 2048 * 2);
  bf16_t* Wuq_t  = (bf16_t*)carve((size_t)2048 * 1024 * 2);
  bf16_t* Wdkv_t = (bf16_t*)carve((size_t)1536 * 2048 * 2);
  bf16_t* Wukv_t = (bf16_t*)carve((size_t)3072 * 1408 * 2);
  bf16_t* Wo_bf  = (bf16_t*)carve((size_t)2048 * 2048 * 2);
  float*  Cq     = (float*)carve(R * 1024 * 4);
  bf16_t* cq_bf  = (bf16_t*)carve(R * 1024 * 2);
  float*  Qf     = (float*)carve(R * 2048 * 4);
  float*  Ckv    = (float*)carve(R * 1536 * 4);
  bf16_t* kvl_bf = (bf16_t*)carve(R * 1408 * 2);
  float*  KVf    = (float*)carve(R * 3072 * 4);
  bf16_t* qh     = (bf16_t*)carve(R * 2048 * 2);
  bf16_t* kh     = (bf16_t*)carve(R * 2048 * 2);
  bf16_t* vt     = (bf16_t*)carve(R * 2048 * 2);
  bf16_t* attn_b = (bf16_t*)carve(R * 2048 * 2);

  dim3 b256(256), b16x16(16, 16);

  // 1) precision conversions / weight pre-transposes (zero-padded)
  k_f32_to_bf16<<<32768, b256, 0, stream>>>(q_input, q_in, R * 2048);
  k_f32_to_bf16<<<32768, b256, 0, stream>>>(kv_input, kv_in, R * 2048);
  k_transpose_bf16<<<dim3(64, 128), b16x16, 0, stream>>>(W_dq, Wdq_t, 2048, 1024, 2048, 1024);
  k_transpose_bf16<<<dim3(128, 64), b16x16, 0, stream>>>(W_uq, Wuq_t, 1024, 2048, 1024, 2048);
  k_transpose_bf16<<<dim3(96, 128), b16x16, 0, stream>>>(W_dkv, Wdkv_t, 2048, 1429, 2048, 1536);
  k_transpose_bf16<<<dim3(192, 88), b16x16, 0, stream>>>(W_ukv, Wukv_t, 1365, 3072, 1408, 3072);
  k_f32_to_bf16<<<16384, b256, 0, stream>>>(W_o, Wo_bf, (size_t)2048 * 2048);

  // 2) Q path: down-proj -> LN -> up-proj -> RoPE/head split
  k_gemm_bf16<<<dim3(8, 32), b256, 0, stream>>>(q_in, Wdq_t, Cq, 4096, 1024, 2048, 2048, 2048, 1024);
  k_layernorm_bf16<<<4096, b256, 0, stream>>>(Cq, 1024, cq_bf, 1024, 1024, 1024);
  k_gemm_bf16<<<dim3(16, 32), b256, 0, stream>>>(cq_bf, Wuq_t, Qf, 4096, 2048, 1024, 1024, 1024, 2048);
  k_assemble_q<<<32768, b256, 0, stream>>>(Qf, cosc, sinc, qh);

  // 3) KV path: down-proj (-> output #2) -> LN -> up-proj -> K/V assembly
  k_gemm_bf16<<<dim3(12, 32), b256, 0, stream>>>(kv_in, Wdkv_t, Ckv, 4096, 1429, 2048, 2048, 2048, 1536);
  k_copy_ckv<<<22864, b256, 0, stream>>>(Ckv, ckv_out);
  k_layernorm_bf16<<<4096, b256, 0, stream>>>(Ckv, 1536, kvl_bf, 1408, 1365, 1408);
  k_gemm_bf16<<<dim3(24, 32), b256, 0, stream>>>(kvl_bf, Wukv_t, KVf, 4096, 3072, 1408, 1408, 1408, 3072);
  k_assemble_kv<<<49152, b256, 0, stream>>>(KVf, Ckv, cosc, sinc, kh, vt);

  // 4) flash attention (TDM-staged K/V, streaming softmax, WMMA everywhere)
  k_flash_attn<<<dim3(16, 16, 2), b256, 0, stream>>>(qh, kh, vt, attn_b);

  // 5) output projection: out = attn @ W_o^T  (W_o row-major == B^T layout)
  k_gemm_bf16<<<dim3(16, 32), b256, 0, stream>>>(attn_b, Wo_bf, out, 4096, 2048, 2048, 2048, 2048, 2048);
}
